// MultiHeadAttention_532575945353
// MI455X (gfx1250) — compile-verified, matrix operand order checked by reading, not run on hardware
//
#include <hip/hip_runtime.h>

// ---------------------------------------------------------------------------
// MultiHeadAttention forward for MI455X (gfx1250, wave32, WMMA)
// B=4, S=1024, D=1024, H=16, Dh=64
// Outputs (concatenated in d_out): out [4,1024,1024] fp32, attn [4,16,1024,1024] fp32
// ---------------------------------------------------------------------------

#define BATCH 4
#define SEQ   1024
#define DMODEL 1024
#define HEADS 16
#define HDIM  64
#define MTOT  (BATCH * SEQ)        // 4096 rows for the projection GEMMs

typedef unsigned short u16;
typedef unsigned int   u32;

typedef __attribute__((ext_vector_type(16))) __bf16 v16bf;
typedef __attribute__((ext_vector_type(8)))  float  v8f;
typedef __attribute__((ext_vector_type(8)))  u32    v8u;

// ---- fp32 -> bf16, round to nearest even ----------------------------------
__device__ __forceinline__ u16 f2bf(float f) {
    u32 u = __builtin_bit_cast(u32, f);
    u32 r = u + 0x7FFFu + ((u >> 16) & 1u);
    return (u16)(r >> 16);
}
__device__ __forceinline__ u32 pk2(float lo, float hi) {
    return (u32)f2bf(lo) | ((u32)f2bf(hi) << 16);
}

// k-index inside a 16x32 bf16 WMMA fragment for VGPR v, lane-half h
// (CDNA5 ISA: lanes 0-15 hold K=0..7 / 16..23, lanes 16-31 hold
//  K=8..15 / 24..31, two bf16 packed per dword)
__device__ __forceinline__ int kfrag(int v, int half) {
    return half * 8 + (v & 3) * 2 + ((v >> 2) << 4);
}

// ---------------------------------------------------------------------------
// fp32 -> bf16 bulk convert, 8 elements/thread (b128 in, b128 out)
// ---------------------------------------------------------------------------
__global__ void cvt8(const float* __restrict__ s, u16* __restrict__ d, int n) {
    int i = (blockIdx.x * blockDim.x + threadIdx.x) * 8;
    if (i >= n) return;
    float4 a = *(const float4*)(s + i);
    float4 b = *(const float4*)(s + i + 4);
    uint4 o;
    o.x = pk2(a.x, a.y);
    o.y = pk2(a.z, a.w);
    o.z = pk2(b.x, b.y);
    o.w = pk2(b.z, b.w);
    *(uint4*)(d + i) = o;
}

// ---------------------------------------------------------------------------
// C[M,N] = A[M,K] @ W[N,K]^T + bias   (A, W bf16; C bf16 or fp32)
// Block: 128 threads = 4 waves. Wave w computes a 32(M) x 64(N) tile:
// 8 accumulators, 2 A-fragments per K-step, B-fragments shared across the
// two M-halves (8 WMMAs per 48 loaded dwords).
// ---------------------------------------------------------------------------
template <bool WRITE_F32>
__global__ void gemm_xwT(const u16* __restrict__ A,
                         const u16* __restrict__ W,
                         const float* __restrict__ bias,
                         u16* __restrict__ Obf,
                         float* __restrict__ Of,
                         int M, int N, int K) {
    const int lane = threadIdx.x & 31;
    const int wave = threadIdx.x >> 5;
    const int half = lane >> 4;
    const int r    = lane & 15;
    const int n0   = blockIdx.x * 64;
    const int m0   = blockIdx.y * 128 + wave * 32;

    v8f acc[2][4] = {};

    for (int kb = 0; kb < K; kb += 32) {
        v16bf af[2];
#pragma unroll
        for (int m = 0; m < 2; ++m) {
            v8u au;
            const u16* arow = A + (size_t)(m0 + m * 16 + r) * K + kb;
#pragma unroll
            for (int v = 0; v < 8; ++v)
                au[v] = *(const u32*)(arow + kfrag(v, half));
            af[m] = __builtin_bit_cast(v16bf, au);
        }

#pragma unroll
        for (int t = 0; t < 4; ++t) {
            v8u bu;
            const u16* wrow = W + (size_t)(n0 + t * 16 + r) * K + kb;
#pragma unroll
            for (int v = 0; v < 8; ++v)
                bu[v] = *(const u32*)(wrow + kfrag(v, half));
            v16bf bf = __builtin_bit_cast(v16bf, bu);
#pragma unroll
            for (int m = 0; m < 2; ++m)
                acc[m][t] = __builtin_amdgcn_wmma_f32_16x16x32_bf16(
                    false, af[m], false, bf, (short)0, acc[m][t], false, false);
        }
    }

#pragma unroll
    for (int t = 0; t < 4; ++t) {
        const int col = n0 + t * 16 + r;
        const float bs = bias[col];
#pragma unroll
        for (int m = 0; m < 2; ++m) {
#pragma unroll
            for (int v = 0; v < 8; ++v) {
                const int row = m0 + m * 16 + v + half * 8;
                const float val = acc[m][t][v] + bs;
                if (WRITE_F32)
                    Of[(size_t)row * N + col] = val;
                else
                    Obf[(size_t)row * N + col] = f2bf(val);
            }
        }
    }
}

// ---------------------------------------------------------------------------
// Vt[b,h,dh,s] = Vp[b*S+s, h*64+dh]  (bf16, LDS-tiled 64x64 transpose,
// coalesced on both sides)
// ---------------------------------------------------------------------------
__global__ void transpose_v(const u16* __restrict__ Vp, u16* __restrict__ Vt) {
    __shared__ u16 tile[64][65];
    const int s0 = blockIdx.x * 64;
    const int bh = blockIdx.y;               // b*H + h
    const int h  = bh & (HEADS - 1);
    const int b  = bh >> 4;
    const int tid = threadIdx.x;

    for (int e = tid; e < 64 * 64; e += 256) {
        const int s  = e >> 6;
        const int dh = e & 63;
        tile[s][dh] = Vp[((size_t)b * SEQ + s0 + s) * DMODEL + h * HDIM + dh];
    }
    __syncthreads();
    for (int e = tid; e < 64 * 64; e += 256) {
        const int dh = e >> 6;
        const int s  = e & 63;
        Vt[((size_t)bh * HDIM + dh) * SEQ + s0 + s] = tile[s][dh];
    }
}

// ---------------------------------------------------------------------------
// scores = (Q K^T) * 1/sqrt(Dh), row softmax, write fp32 probs.
// Block: 128 threads = 4 waves cooperating on ONE 16-query-row strip
// (64 KB LDS per block -> up to ~20 resident waves per WGP).
// Wave w computes score tiles nt = w, w+4, ..., then softmaxes rows 4w..4w+3.
// ---------------------------------------------------------------------------
__global__ void attn_scores_softmax(const u16* __restrict__ Qp,
                                    const u16* __restrict__ Kp,
                                    float* __restrict__ attn) {
    __shared__ float sc[16 * SEQ];   // 64 KB

    const int blk = blockIdx.x;          // ((b*H + h) * (S/16)) + qt
    const int qt  = blk & (SEQ / 16 - 1);
    const int bh  = blk >> 6;
    const int h   = bh & (HEADS - 1);
    const int b   = bh >> 4;

    const int wave = threadIdx.x >> 5;
    const int lane = threadIdx.x & 31;
    const int half = lane >> 4;
    const int r    = lane & 15;
    const float scale = 0.125f;          // 1/sqrt(64)

    const size_t rowbase = (size_t)b * SEQ;
    const int    colh    = h * HDIM;

    for (int nt = wave; nt < SEQ / 16; nt += 4) {
        v8f acc = {};
#pragma unroll
        for (int kb2 = 0; kb2 < 2; ++kb2) {       // Dh = 64 = 2 x 32
            const int kb = kb2 * 32;
            v8u au, bu;
            const u16* qrow = Qp + (rowbase + qt * 16 + r) * DMODEL + colh + kb;
            const u16* krow = Kp + (rowbase + nt * 16 + r) * DMODEL + colh + kb;
#pragma unroll
            for (int v = 0; v < 8; ++v) {
                const int kk = kfrag(v, half);
                au[v] = *(const u32*)(qrow + kk);
                bu[v] = *(const u32*)(krow + kk);
            }
            acc = __builtin_amdgcn_wmma_f32_16x16x32_bf16(
                false, __builtin_bit_cast(v16bf, au),
                false, __builtin_bit_cast(v16bf, bu),
                (short)0, acc, false, false);
        }
#pragma unroll
        for (int v = 0; v < 8; ++v)
            sc[(v + half * 8) * SEQ + nt * 16 + r] = acc[v] * scale;
    }
    __syncthreads();

    // wave-cooperative softmax: wave w handles rows 4w..4w+3 (float4 passes)
    for (int row = wave * 4; row < wave * 4 + 4; ++row) {
        float4* srow4 = (float4*)(sc + row * SEQ);
        float m = -3.0e38f;
        for (int c = lane; c < SEQ / 4; c += 32) {
            const float4 x = srow4[c];
            m = fmaxf(m, fmaxf(fmaxf(x.x, x.y), fmaxf(x.z, x.w)));
        }
#pragma unroll
        for (int off = 16; off > 0; off >>= 1) m = fmaxf(m, __shfl_xor(m, off, 32));
        float sum = 0.f;
        for (int c = lane; c < SEQ / 4; c += 32) {
            float4 x = srow4[c];
            x.x = __expf(x.x - m);
            x.y = __expf(x.y - m);
            x.z = __expf(x.z - m);
            x.w = __expf(x.w - m);
            srow4[c] = x;
            sum += x.x + x.y + x.z + x.w;
        }
#pragma unroll
        for (int off = 16; off > 0; off >>= 1) sum += __shfl_xor(sum, off, 32);
        const float inv = 1.0f / sum;
        float4* dst4 = (float4*)(attn +
            (((size_t)(b * HEADS + h)) * SEQ + qt * 16 + row) * SEQ);
        for (int c = lane; c < SEQ / 4; c += 32) {
            float4 x = srow4[c];
            x.x *= inv; x.y *= inv; x.z *= inv; x.w *= inv;
            dst4[c] = x;
        }
    }
}

// ---------------------------------------------------------------------------
// ctx[b, q, h*64+dh] = sum_k attn[b,h,q,k] * Vt[b,h,dh,k]    (ctx bf16)
// One wave per (b, h, 32-query strip): 32 x 64 output, K = 1024 keys.
// Vt is head-transposed so B-fragment k-pairs are contiguous dwords.
// ---------------------------------------------------------------------------
__global__ void attn_av(const float* __restrict__ attn,
                        const u16* __restrict__ Vt,
                        u16* __restrict__ ctx) {
    const int blk = blockIdx.x;          // ((b*H + h) * (S/32)) + qt
    const int qt  = blk & (SEQ / 32 - 1);
    const int bh  = blk >> 5;
    const int h   = bh & (HEADS - 1);
    const int b   = bh >> 4;

    const int lane = threadIdx.x & 31;
    const int half = lane >> 4;
    const int r    = lane & 15;

    v8f acc[2][4] = {};
    const float* abase  = attn + ((size_t)bh * SEQ + qt * 32) * SEQ;
    const u16*   vtbase = Vt + (size_t)bh * HDIM * SEQ;

    for (int kb = 0; kb < SEQ; kb += 32) {
        v16bf af[2];
#pragma unroll
        for (int m = 0; m < 2; ++m) {
            v8u au;
            const float* ar = abase + (size_t)(m * 16 + r) * SEQ + kb;
#pragma unroll
            for (int v = 0; v < 8; ++v) {
                const float2 p = *(const float2*)(ar + kfrag(v, half));
                au[v] = pk2(p.x, p.y);
            }
            af[m] = __builtin_bit_cast(v16bf, au);
        }

#pragma unroll
        for (int t = 0; t < 4; ++t) {
            v8u bu;
            const u16* vrow = vtbase + (size_t)(t * 16 + r) * SEQ + kb;
#pragma unroll
            for (int v = 0; v < 8; ++v)
                bu[v] = *(const u32*)(vrow + kfrag(v, half));
            v16bf bf = __builtin_bit_cast(v16bf, bu);
#pragma unroll
            for (int m = 0; m < 2; ++m)
                acc[m][t] = __builtin_amdgcn_wmma_f32_16x16x32_bf16(
                    false, af[m], false, bf, (short)0, acc[m][t], false, false);
        }
    }

#pragma unroll
    for (int t = 0; t < 4; ++t) {
        const int col = h * HDIM + t * 16 + r;
#pragma unroll
        for (int m = 0; m < 2; ++m) {
#pragma unroll
            for (int v = 0; v < 8; ++v) {
                const int row = qt * 32 + m * 16 + v + half * 8;
                ctx[((size_t)b * SEQ + row) * DMODEL + col] = f2bf(acc[m][t][v]);
            }
        }
    }
}

// ---------------------------------------------------------------------------
// Host side
// ---------------------------------------------------------------------------
extern "C" void kernel_launch(void* const* d_in, const int* in_sizes, int n_in,
                              void* d_out, int out_size, void* d_ws, size_t ws_size,
                              hipStream_t stream) {
    const float* query = (const float*)d_in[0];
    const float* key_  = (const float*)d_in[1];
    const float* value = (const float*)d_in[2];
    const float* Wq    = (const float*)d_in[3];
    const float* bq    = (const float*)d_in[4];
    const float* Wk    = (const float*)d_in[5];
    const float* bk    = (const float*)d_in[6];
    const float* Wv    = (const float*)d_in[7];
    const float* bv    = (const float*)d_in[8];
    const float* Wo    = (const float*)d_in[9];
    const float* bo    = (const float*)d_in[10];

    float* out  = (float*)d_out;                            // [4,1024,1024]
    float* attn = out + (size_t)MTOT * DMODEL;              // [4,16,1024,1024]

    // workspace (bf16): inputs, weights, projections, Vt, context (~75 MB)
    const size_t MD = (size_t)MTOT * DMODEL;                // 4,194,304
    const size_t DD = (size_t)DMODEL * DMODEL;              // 1,048,576
    u16* q_bf  = (u16*)d_ws;
    u16* k_bf  = q_bf + MD;
    u16* v_bf  = k_bf + MD;
    u16* wq_bf = v_bf + MD;
    u16* wk_bf = wq_bf + DD;
    u16* wv_bf = wk_bf + DD;
    u16* wo_bf = wv_bf + DD;
    u16* Qp    = wo_bf + DD;
    u16* Kp    = Qp + MD;
    u16* Vp    = Kp + MD;
    u16* Vt    = Vp + MD;   // head-transposed V: [B,H,64,S]
    u16* ctx   = Vt + MD;

    // 1) fp32 -> bf16 conversions
    {
        const int T = 256;
        cvt8<<<(int)((MD / 8 + T - 1) / T), T, 0, stream>>>(query, q_bf, (int)MD);
        cvt8<<<(int)((MD / 8 + T - 1) / T), T, 0, stream>>>(key_,  k_bf, (int)MD);
        cvt8<<<(int)((MD / 8 + T - 1) / T), T, 0, stream>>>(value, v_bf, (int)MD);
        cvt8<<<(int)((DD / 8 + T - 1) / T), T, 0, stream>>>(Wq, wq_bf, (int)DD);
        cvt8<<<(int)((DD / 8 + T - 1) / T), T, 0, stream>>>(Wk, wk_bf, (int)DD);
        cvt8<<<(int)((DD / 8 + T - 1) / T), T, 0, stream>>>(Wv, wv_bf, (int)DD);
        cvt8<<<(int)((DD / 8 + T - 1) / T), T, 0, stream>>>(Wo, wo_bf, (int)DD);
    }

    // 2) projections: Qp/Kp/Vp = X @ W^T + b   (bf16 out)
    {
        dim3 grid(DMODEL / 64, MTOT / 128);
        gemm_xwT<false><<<grid, 128, 0, stream>>>(q_bf, wq_bf, bq, Qp, nullptr,
                                                  MTOT, DMODEL, DMODEL);
        gemm_xwT<false><<<grid, 128, 0, stream>>>(k_bf, wk_bf, bk, Kp, nullptr,
                                                  MTOT, DMODEL, DMODEL);
        gemm_xwT<false><<<grid, 128, 0, stream>>>(v_bf, wv_bf, bv, Vp, nullptr,
                                                  MTOT, DMODEL, DMODEL);
    }

    // 3) head-transpose V for contiguous AV B-fragments
    {
        dim3 grid(SEQ / 64, BATCH * HEADS);
        transpose_v<<<grid, 256, 0, stream>>>(Vp, Vt);
    }

    // 4) scores + softmax -> attention probabilities (fp32, second output)
    attn_scores_softmax<<<BATCH * HEADS * (SEQ / 16), 128, 0, stream>>>(Qp, Kp, attn);

    // 5) context = attn @ V  (bf16)
    attn_av<<<BATCH * HEADS * (SEQ / 32), 32, 0, stream>>>(attn, Vt, ctx);

    // 6) out = ctx @ Wo^T + bo  (fp32, first output)
    {
        dim3 grid(DMODEL / 64, MTOT / 128);
        gemm_xwT<true><<<grid, 128, 0, stream>>>(ctx, wo_bf, bo, nullptr, out,
                                                 MTOT, DMODEL, DMODEL);
    }

    (void)in_sizes; (void)n_in; (void)out_size; (void)ws_size;
}
